// GraphTransformerLayer_13460427506263
// MI455X (gfx1250) — matrix operand order repaired and checked by reading, not compiled, not run on hardware
//
#include <hip/hip_runtime.h>
#include <hip/hip_bf16.h>

// ---------------------------------------------------------------------------
// CDNA5 (gfx1250) wave32 WMMA types & helpers
// ---------------------------------------------------------------------------
typedef __attribute__((ext_vector_type(16))) __bf16 v16bf;
typedef __attribute__((ext_vector_type(8)))  float  v8f;

static __device__ inline v8f wmma_bf16(const v16bf& a, const v16bf& b, const v8f& c) {
    // D = A(16x32 bf16) * B(32x16 bf16) + C(16x16 f32)
    return __builtin_amdgcn_wmma_f32_16x16x32_bf16(false, a, false, b, (short)0, c, false, false);
}

// A-fragment K index for (vgpr pair v, lane-group g):  ISA 7.12.2 16-bit A 16x32
// lanes 0-15 (g=0): V0..3 -> K=0..7, V4..7 -> K=16..23 ; lanes 16-31 (g=1): +8
static __device__ inline int a_kbase(int v, int g) {
    return (v < 4) ? (8 * g + 2 * v) : (8 + 8 * g + 2 * v); // v>=4 => 16 + 8g + 2(v-4)
}
// B-fragment K index: same as A: V0..3 -> K=8g..8g+7, V4..7 -> K=16+8g..16+8g+7
static __device__ inline int b_kbase(int v, int g) { return (v < 4) ? (8 * g + 2 * v) : (8 + 8 * g + 2 * v); }

static __device__ inline float leaky02(float x) { return x < 0.0f ? 0.2f * x : x; }

union BFragU { v16bf bf; unsigned int u[8]; };

// ---------------------------------------------------------------------------
// Kernel 0: repack eb_w1 [256,64] f32 into B-fragment dword layout (bf16 pairs)
// index = ((ntile*8 + kstep)*8 + v)*32 + lane
// ---------------------------------------------------------------------------
__global__ void prepack_w1_kernel(const float* __restrict__ W1, unsigned int* __restrict__ P) {
    int idx   = blockIdx.x * 256 + threadIdx.x;   // 0..8191
    int lane  = idx & 31;
    int v     = (idx >> 5) & 7;
    int kstep = (idx >> 8) & 7;
    int ntile = (idx >> 11) & 3;
    int g  = lane >> 4;
    int ln = lane & 15;
    int n  = ntile * 16 + ln;
    int k  = kstep * 32 + b_kbase(v, g);
    __bf16 lo = (__bf16)W1[(size_t)k * 64 + n];
    __bf16 hi = (__bf16)W1[(size_t)(k + 1) * 64 + n];
    unsigned short ulo, uhi;
    __builtin_memcpy(&ulo, &lo, 2);
    __builtin_memcpy(&uhi, &hi, 2);
    P[idx] = ((unsigned int)uhi << 16) | (unsigned int)ulo;
}

// ---------------------------------------------------------------------------
// Kernel 1: generic GEMM  C[M,N] = act(A[M,K] @ W[K,N] + bias)
// block = 256 threads (8 waves), tile = 64 rows x 128 cols, K chunks of 64.
// act: 0 = none, 1 = exact GELU
// ---------------------------------------------------------------------------
__global__ __launch_bounds__(256) void gemm_bias_act_kernel(
    const float* __restrict__ A, const float* __restrict__ W,
    const float* __restrict__ bias, float* __restrict__ C,
    int M, int K, int N, int act)
{
    __shared__ __bf16 Albs[64][72];    // [row][k-in-chunk]
    __shared__ __bf16 Wlds[128][72];   // [n][k-in-chunk] (transposed)

    const int mBase = blockIdx.x * 64;
    const int nBase = blockIdx.y * 128;
    const int t    = threadIdx.x;
    const int lane = t & 31;
    const int w    = t >> 5;
    const int g    = lane >> 4;
    const int ln   = lane & 15;
    const int mtile = w & 3;          // 0..3 (16-row tiles)
    const int ngrp  = (w >> 2) * 4;   // ntiles ngrp..ngrp+3

    v8f acc[4] = {v8f{}, v8f{}, v8f{}, v8f{}};

    const int arow_ld = t >> 2;
    const int ak0     = (t & 3) * 16;
    const int wk_ld   = t >> 2;
    const int wn0     = (t & 3) * 32;

    for (int kc = 0; kc < K; kc += 64) {
        __syncthreads();
        // stage A chunk (f32 -> bf16)
        {
            const float* src = A + (size_t)(mBase + arow_ld) * K + kc + ak0;
            #pragma unroll
            for (int i = 0; i < 16; i += 4) {
                float4 f = *(const float4*)(src + i);
                Albs[arow_ld][ak0 + i + 0] = (__bf16)f.x;
                Albs[arow_ld][ak0 + i + 1] = (__bf16)f.y;
                Albs[arow_ld][ak0 + i + 2] = (__bf16)f.z;
                Albs[arow_ld][ak0 + i + 3] = (__bf16)f.w;
            }
        }
        // stage W chunk transposed (f32 -> bf16)
        {
            const float* src = W + (size_t)(kc + wk_ld) * N + nBase + wn0;
            #pragma unroll
            for (int i = 0; i < 32; ++i) Wlds[wn0 + i][wk_ld] = (__bf16)src[i];
        }
        if (kc + 64 < K) { // hint next A chunk into cache (global_prefetch_b8)
            __builtin_prefetch(A + (size_t)(mBase + arow_ld) * K + kc + 64 + ak0, 0, 0);
        }
        __syncthreads();

        #pragma unroll
        for (int ks = 0; ks < 2; ++ks) {
            // load A fragment
            v16bf afrag;
            const int arow = mtile * 16 + ln;
            #pragma unroll
            for (int vv = 0; vv < 8; ++vv) {
                int kb = ks * 32 + a_kbase(vv, g);
                afrag[2 * vv]     = Albs[arow][kb];
                afrag[2 * vv + 1] = Albs[arow][kb + 1];
            }
            // load ALL B fragments into distinct registers first ...
            v16bf bfrag[4];
            #pragma unroll
            for (int nt = 0; nt < 4; ++nt) {
                const int ncol = (ngrp + nt) * 16 + ln;
                #pragma unroll
                for (int vv = 0; vv < 8; ++vv) {
                    int kb = ks * 32 + b_kbase(vv, g);
                    bfrag[nt][2 * vv]     = Wlds[ncol][kb];
                    bfrag[nt][2 * vv + 1] = Wlds[ncol][kb + 1];
                }
            }
            // ... then fire independent WMMAs back-to-back (XDL pipelining)
            #pragma unroll
            for (int nt = 0; nt < 4; ++nt)
                acc[nt] = wmma_bf16(afrag, bfrag[nt], acc[nt]);
        }
    }

    // epilogue: bias + activation + store
    #pragma unroll
    for (int nt = 0; nt < 4; ++nt) {
        const int ncol = nBase + (ngrp + nt) * 16 + ln;
        const float bv = bias ? bias[ncol] : 0.0f;
        #pragma unroll
        for (int r = 0; r < 8; ++r) {
            const int row = mBase + mtile * 16 + r + 8 * g;
            float x = acc[nt][r] + bv;
            if (act == 1) x = 0.5f * x * (1.0f + erff(x * 0.70710678118654752f));
            C[(size_t)row * N + ncol] = x;
        }
    }
}

// ---------------------------------------------------------------------------
// Kernel 2: fused edge-bias MLP + SPD bias + mask select
//   rows = B*N*N flattened (j fastest); 64 rows per block; 8192 blocks.
//   bias_out layout: [B, H=8, N, N]
// ---------------------------------------------------------------------------
__global__ __launch_bounds__(256) void edge_bias_kernel(
    const float* __restrict__ e, const unsigned int* __restrict__ W1p,
    const float* __restrict__ eb_b1, const float* __restrict__ eb_g,
    const float* __restrict__ eb_beta, const float* __restrict__ eb_w2,
    const float* __restrict__ eb_b2,
    const float* __restrict__ spd, const float* __restrict__ spd_w1,
    const float* __restrict__ spd_b1, const float* __restrict__ spd_w2,
    const float* __restrict__ spd_b2,
    const int* __restrict__ edge_mask, float* __restrict__ bias_out)
{
    __shared__ __bf16 elds[64][264];   // e tile, bf16 (pad avoids bank conflicts)
    __shared__ float  X[64][65];       // e@W1 + b1
    __shared__ __bf16 Xn[64][66];      // LN + leaky, bf16
    __shared__ float  s_b1[64], s_g[64], s_beta[64], s_w2[512], s_b2[8];
    __shared__ float  s_sw1[64], s_sb1[64], s_sw2[512], s_sb2[8];

    const int t    = threadIdx.x;
    const int lane = t & 31;
    const int w    = t >> 5;
    const int g    = lane >> 4;
    const int ln   = lane & 15;
    const size_t rowBase = (size_t)blockIdx.x * 64;

    if (t < 64) {
        s_b1[t] = eb_b1[t]; s_g[t] = eb_g[t]; s_beta[t] = eb_beta[t];
        s_sw1[t] = spd_w1[t]; s_sb1[t] = spd_b1[t];
    }
    if (t < 8) { s_b2[t] = eb_b2[t]; s_sb2[t] = spd_b2[t]; }
    for (int i = t; i < 512; i += 256) { s_w2[i] = eb_w2[i]; s_sw2[i] = spd_w2[i]; }

    // stage e tile: 64 rows x 256 k, f32 -> bf16
    {
        const int row = t >> 2;
        const int k0  = (t & 3) * 64;
        const float* src = e + (rowBase + row) * 256 + k0;
        #pragma unroll
        for (int i = 0; i < 64; i += 4) {
            float4 f = *(const float4*)(src + i);
            elds[row][k0 + i + 0] = (__bf16)f.x;
            elds[row][k0 + i + 1] = (__bf16)f.y;
            elds[row][k0 + i + 2] = (__bf16)f.z;
            elds[row][k0 + i + 3] = (__bf16)f.w;
        }
    }
    __syncthreads();

    // GEMM1: [64,256] @ [256,64] via WMMA; wave w -> mtile=w&3, ntiles=(w>>2)*2+{0,1}
    const int mtile = w & 3;
    const int np    = (w >> 2) * 2;
    v8f acc[2] = {v8f{}, v8f{}};
    #pragma unroll
    for (int ks = 0; ks < 8; ++ks) {
        v16bf afrag;
        const int arow = mtile * 16 + ln;
        #pragma unroll
        for (int vv = 0; vv < 8; ++vv) {
            int kb = ks * 32 + a_kbase(vv, g);
            afrag[2 * vv]     = elds[arow][kb];
            afrag[2 * vv + 1] = elds[arow][kb + 1];
        }
        // both B fragments first (coalesced dword loads from prepacked L2 copy)
        BFragU bu[2];
        #pragma unroll
        for (int p = 0; p < 2; ++p) {
            const int base = ((np + p) * 8 + ks) * 256 + lane;
            #pragma unroll
            for (int vv = 0; vv < 8; ++vv) bu[p].u[vv] = W1p[base + vv * 32];
        }
        #pragma unroll
        for (int p = 0; p < 2; ++p)
            acc[p] = wmma_bf16(afrag, bu[p].bf, acc[p]);
    }
    #pragma unroll
    for (int p = 0; p < 2; ++p) {
        const int ncol = (np + p) * 16 + ln;
        #pragma unroll
        for (int r = 0; r < 8; ++r)
            X[mtile * 16 + r + 8 * g][ncol] = acc[p][r] + s_b1[ncol];
    }
    __syncthreads();

    // LayerNorm over 64 + LeakyReLU(0.2): 4 threads per row, 16 elems each
    {
        const int row = t >> 2;
        const int q   = t & 3;
        float vals[16];
        float sum = 0.0f;
        #pragma unroll
        for (int i = 0; i < 16; ++i) { vals[i] = X[row][q * 16 + i]; sum += vals[i]; }
        sum += __shfl_xor(sum, 1, 32);
        sum += __shfl_xor(sum, 2, 32);
        const float mean = sum * (1.0f / 64.0f);
        float sq = 0.0f;
        #pragma unroll
        for (int i = 0; i < 16; ++i) { float d = vals[i] - mean; sq += d * d; }
        sq += __shfl_xor(sq, 1, 32);
        sq += __shfl_xor(sq, 2, 32);
        const float rstd = rsqrtf(sq * (1.0f / 64.0f) + 1e-5f);
        #pragma unroll
        for (int i = 0; i < 16; ++i) {
            const int c = q * 16 + i;
            float y = (vals[i] - mean) * rstd * s_g[c] + s_beta[c];
            Xn[row][c] = (__bf16)leaky02(y);
        }
    }
    __syncthreads();

    // GEMM2 [64,64]@[64,8] (VALU, tiny) + SPD MLP + mask select + scatter store
    {
        const int row = t >> 1;
        const int hh  = (t & 1) * 4;
        const size_t R = rowBase + row;
        const int b = (int)(R >> 16);
        const int i = (int)((R >> 8) & 255);
        const int j = (int)(R & 255);

        float sv = spd[R];
        sv = fminf(fmaxf(sv, 0.0f), 10.0f) * 0.1f;

        float accE[4], accS[4];
        #pragma unroll
        for (int c = 0; c < 4; ++c) { accE[c] = s_b2[hh + c]; accS[c] = s_sb2[hh + c]; }
        for (int k = 0; k < 64; ++k) {
            const float x   = (float)Xn[row][k];
            const float hsp = leaky02(sv * s_sw1[k] + s_sb1[k]);
            #pragma unroll
            for (int c = 0; c < 4; ++c) {
                accE[c] += x   * s_w2[k * 8 + hh + c];
                accS[c] += hsp * s_sw2[k * 8 + hh + c];
            }
        }
        const int em = edge_mask[R];
        #pragma unroll
        for (int c = 0; c < 4; ++c) {
            const float v = em ? accE[c] : accS[c];
            bias_out[(((size_t)b * 8 + hh + c) * 256 + i) * 256 + j] = v;
        }
    }
}

// ---------------------------------------------------------------------------
// Kernel 3: fused attention per (b,h, 32-row i-tile)
//   scores = Q K^T / sqrt(32) + bias, node-mask, softmax, out = attn @ V
// ---------------------------------------------------------------------------
__global__ __launch_bounds__(256) void attn_kernel(
    const float* __restrict__ Q, const float* __restrict__ Kmat,
    const float* __restrict__ V, const float* __restrict__ bias,
    const int* __restrict__ node_mask, float* __restrict__ out)
{
    __shared__ float sc[32][260];

    const int bh = blockIdx.x;        // b*8+h
    const int b  = bh >> 3;
    const int h  = bh & 7;
    const int ib = blockIdx.y;        // i-tile (32 rows)
    const int t    = threadIdx.x;
    const int lane = t & 31;
    const int w    = t >> 5;
    const int g    = lane >> 4;
    const int ln   = lane & 15;
    const float scale = 0.17677669529663687f; // 1/sqrt(32)

    // ---- phase 1: scores ----
    {
        const int mtile = w & 1;
        const int ng    = (w >> 1) * 4;
        const int iRow  = ib * 32 + mtile * 16 + ln;
        const float* qrow = Q + ((size_t)(b * 256 + iRow)) * 256 + h * 32;
        v16bf afrag;
        #pragma unroll
        for (int vv = 0; vv < 8; ++vv) {
            int kb = a_kbase(vv, g);
            afrag[2 * vv]     = (__bf16)qrow[kb];
            afrag[2 * vv + 1] = (__bf16)qrow[kb + 1];
        }
        // all four K fragments first, then back-to-back WMMAs
        v16bf bfrag[4];
        #pragma unroll
        for (int nt = 0; nt < 4; ++nt) {
            const int j = (ng + nt) * 16 + ln;
            const float* krow = Kmat + ((size_t)(b * 256 + j)) * 256 + h * 32;
            #pragma unroll
            for (int vv = 0; vv < 8; ++vv) {
                int kb = b_kbase(vv, g);
                bfrag[nt][2 * vv]     = (__bf16)krow[kb];
                bfrag[nt][2 * vv + 1] = (__bf16)krow[kb + 1];
            }
        }
        v8f acc[4] = {v8f{}, v8f{}, v8f{}, v8f{}};
        #pragma unroll
        for (int nt = 0; nt < 4; ++nt)
            acc[nt] = wmma_bf16(afrag, bfrag[nt], acc[nt]);
        #pragma unroll
        for (int nt = 0; nt < 4; ++nt) {
            const int j = (ng + nt) * 16 + ln;
            const float nm = node_mask[b * 256 + j] ? 0.0f : 1.0f;
            #pragma unroll
            for (int r = 0; r < 8; ++r) {
                const int il = mtile * 16 + r + 8 * g;
                const int iG = ib * 32 + il;
                float s = acc[nt][r] * scale +
                          bias[(((size_t)(b * 8 + h)) * 256 + iG) * 256 + j];
                if (nm != 0.0f) s = -1e9f;
                sc[il][j] = s;
            }
        }
    }
    __syncthreads();

    // ---- phase 2: softmax over j (row-wise); 8 threads/row, 32 elems each ----
    {
        const int row = t >> 3;
        const int seg = t & 7;
        float m = -3.4e38f;
        #pragma unroll
        for (int i = 0; i < 32; ++i) m = fmaxf(m, sc[row][seg * 32 + i]);
        m = fmaxf(m, __shfl_xor(m, 1, 32));
        m = fmaxf(m, __shfl_xor(m, 2, 32));
        m = fmaxf(m, __shfl_xor(m, 4, 32));
        float s = 0.0f;
        #pragma unroll
        for (int i = 0; i < 32; ++i) {
            float ev = __expf(sc[row][seg * 32 + i] - m);
            sc[row][seg * 32 + i] = ev;
            s += ev;
        }
        s += __shfl_xor(s, 1, 32);
        s += __shfl_xor(s, 2, 32);
        s += __shfl_xor(s, 4, 32);
        const float inv = 1.0f / s;
        #pragma unroll
        for (int i = 0; i < 32; ++i) sc[row][seg * 32 + i] *= inv;
    }
    __syncthreads();

    // ---- phase 3: out = attn @ V  (waves 0..3; wave-uniform branch) ----
    if (w < 4) {
        const int mtile = w & 1;
        const int ntile = w >> 1;
        const int d = ntile * 16 + ln;
        v8f acc = {};
        for (int ks = 0; ks < 8; ++ks) {
            v16bf afrag;
            const int arow = mtile * 16 + ln;
            #pragma unroll
            for (int vv = 0; vv < 8; ++vv) {
                int kb = ks * 32 + a_kbase(vv, g);
                afrag[2 * vv]     = (__bf16)sc[arow][kb];
                afrag[2 * vv + 1] = (__bf16)sc[arow][kb + 1];
            }
            v16bf bfrag;
            #pragma unroll
            for (int vv = 0; vv < 8; ++vv) {
                const int kj = ks * 32 + b_kbase(vv, g);
                bfrag[2 * vv]     = (__bf16)V[((size_t)(b * 256 + kj)) * 256 + h * 32 + d];
                bfrag[2 * vv + 1] = (__bf16)V[((size_t)(b * 256 + kj + 1)) * 256 + h * 32 + d];
            }
            acc = wmma_bf16(afrag, bfrag, acc);
        }
        #pragma unroll
        for (int r = 0; r < 8; ++r) {
            const int iG = ib * 32 + mtile * 16 + r + 8 * g;
            out[((size_t)(b * 256 + iG)) * 256 + h * 32 + d] = acc[r];
        }
    }
}

// ---------------------------------------------------------------------------
// Kernel 4: out = LayerNorm(res + x) * g + b   (wave per 256-wide row)
// ---------------------------------------------------------------------------
__global__ __launch_bounds__(256) void ln_residual_kernel(
    const float* __restrict__ x, const float* __restrict__ res,
    const float* __restrict__ gm, const float* __restrict__ bt,
    float* __restrict__ out)
{
    const int row  = blockIdx.x * 8 + (threadIdx.x >> 5);
    const int lane = threadIdx.x & 31;
    const float* xr = x   + (size_t)row * 256;
    const float* rr = res + (size_t)row * 256;

    float val[8];
    float sum = 0.0f;
    #pragma unroll
    for (int i = 0; i < 8; ++i) {
        const int c = lane + i * 32;
        val[i] = xr[c] + rr[c];
        sum += val[i];
    }
    #pragma unroll
    for (int m = 1; m < 32; m <<= 1) sum += __shfl_xor(sum, m, 32);
    const float mean = sum * (1.0f / 256.0f);
    float sq = 0.0f;
    #pragma unroll
    for (int i = 0; i < 8; ++i) { float d = val[i] - mean; sq += d * d; }
    #pragma unroll
    for (int m = 1; m < 32; m <<= 1) sq += __shfl_xor(sq, m, 32);
    const float rstd = rsqrtf(sq * (1.0f / 256.0f) + 1e-5f);
    #pragma unroll
    for (int i = 0; i < 8; ++i) {
        const int c = lane + i * 32;
        out[(size_t)row * 256 + c] = (val[i] - mean) * rstd * gm[c] + bt[c];
    }
}

// ---------------------------------------------------------------------------
// Launch
// ---------------------------------------------------------------------------
extern "C" void kernel_launch(void* const* d_in, const int* in_sizes, int n_in,
                              void* d_out, int out_size, void* d_ws, size_t ws_size,
                              hipStream_t stream) {
    (void)in_sizes; (void)n_in; (void)out_size; (void)ws_size;

    const float* h         = (const float*)d_in[0];
    const float* e         = (const float*)d_in[1];
    const int*   node_mask = (const int*)  d_in[2];
    const int*   edge_mask = (const int*)  d_in[3];
    const float* spd       = (const float*)d_in[4];
    const float* wq        = (const float*)d_in[5];
    const float* wk        = (const float*)d_in[6];
    const float* wv        = (const float*)d_in[7];
    const float* eb_w1     = (const float*)d_in[8];
    const float* eb_b1     = (const float*)d_in[9];
    const float* eb_g      = (const float*)d_in[10];
    const float* eb_beta   = (const float*)d_in[11];
    const float* eb_w2     = (const float*)d_in[12];
    const float* eb_b2     = (const float*)d_in[13];
    const float* spd_w1    = (const float*)d_in[14];
    const float* spd_b1    = (const float*)d_in[15];
    const float* spd_w2    = (const float*)d_in[16];
    const float* spd_b2    = (const float*)d_in[17];
    const float* wo_w      = (const float*)d_in[18];
    const float* wo_b      = (const float*)d_in[19];
    const float* ln1_g     = (const float*)d_in[20];
    const float* ln1_b     = (const float*)d_in[21];
    const float* ln2_g     = (const float*)d_in[22];
    const float* ln2_b     = (const float*)d_in[23];
    const float* ffn_w1    = (const float*)d_in[24];
    const float* ffn_b1    = (const float*)d_in[25];
    const float* ffn_w2    = (const float*)d_in[26];
    const float* ffn_b2    = (const float*)d_in[27];

    float* ws = (float*)d_ws;
    size_t off = 0;
    float* qb     = ws + off; off += 2048 * 256;        // Q
    float* kb     = ws + off; off += 2048 * 256;        // K
    float* vb     = ws + off; off += 2048 * 256;        // V
    float* biasb  = ws + off; off += 8 * 8 * 256 * 256; // bias [B,H,N,N]
    float* attnb  = ws + off; off += 2048 * 256;        // attention output
    float* wob    = ws + off; off += 2048 * 256;        // wo projection
    float* h1b    = ws + off; off += 2048 * 256;        // post-LN1
    float* midb   = ws + off; off += 2048 * 1024;       // FFN hidden
    float* ffn2b  = ws + off; off += 2048 * 256;        // FFN out
    unsigned int* w1p = (unsigned int*)(ws + off);      // packed eb_w1 (8192 dwords)

    // 0) repack eb_w1 into WMMA B-fragment layout
    prepack_w1_kernel<<<dim3(32), dim3(256), 0, stream>>>(eb_w1, w1p);

    // 1) QKV projections  [2048,256] @ [256,256]
    gemm_bias_act_kernel<<<dim3(32, 2), dim3(256), 0, stream>>>(h, wq, nullptr, qb, 2048, 256, 256, 0);
    gemm_bias_act_kernel<<<dim3(32, 2), dim3(256), 0, stream>>>(h, wk, nullptr, kb, 2048, 256, 256, 0);
    gemm_bias_act_kernel<<<dim3(32, 2), dim3(256), 0, stream>>>(h, wv, nullptr, vb, 2048, 256, 256, 0);

    // 2) fused edge-bias MLP + SPD + mask select (the ~0.54 GB HBM stream)
    edge_bias_kernel<<<dim3(8192), dim3(256), 0, stream>>>(
        e, w1p, eb_b1, eb_g, eb_beta, eb_w2, eb_b2,
        spd, spd_w1, spd_b1, spd_w2, spd_b2, edge_mask, biasb);

    // 3) fused attention
    attn_kernel<<<dim3(64, 8), dim3(256), 0, stream>>>(qb, kb, vb, biasb, node_mask, attnb);

    // 4) output projection + LN1
    gemm_bias_act_kernel<<<dim3(32, 2), dim3(256), 0, stream>>>(attnb, wo_w, wo_b, wob, 2048, 256, 256, 0);
    ln_residual_kernel<<<dim3(256), dim3(256), 0, stream>>>(wob, h, ln1_g, ln1_b, h1b);

    // 5) FFN (GELU) + LN2
    gemm_bias_act_kernel<<<dim3(32, 8), dim3(256), 0, stream>>>(h1b, ffn_w1, ffn_b1, midb, 2048, 256, 1024, 1);
    gemm_bias_act_kernel<<<dim3(32, 2), dim3(256), 0, stream>>>(midb, ffn_w2, ffn_b2, ffn2b, 2048, 1024, 256, 0);
    ln_residual_kernel<<<dim3(256), dim3(256), 0, stream>>>(ffn2b, h1b, ln2_g, ln2_b, (float*)d_out);
}
